// SemiseparableLayer_83777632076059
// MI455X (gfx1250) — matrix operand order repaired and checked by reading, not compiled, not run on hardware
//
#include <hip/hip_runtime.h>

typedef _Float16 v16h __attribute__((ext_vector_type(16)));
typedef _Float16 v4h  __attribute__((ext_vector_type(4)));
typedef _Float16 v2h  __attribute__((ext_vector_type(2)));
typedef float    v8f  __attribute__((ext_vector_type(8)));

#define N_STAGES 8
#define DI       512
#define DOUT     512
#define SDIM     256
#define BATCH    8192
#define INSZ     4096
#define COLS     64           // batch columns per workgroup
#define NT       4            // 16-wide column tiles per workgroup
#define MT_FULL  48           // (512 y + 256 state) / 16
#define KT_FULL  24           // (256 state + 512 u) / 32
#define TILE_HALFS 512        // 16x32 f16 tile

struct WPtrs {
  const float* A[8]; const float* B[8]; const float* C[8]; const float* D[8];
  const float* E[8]; const float* F[8]; const float* G[8];
};

// ---------------------------------------------------------------------------
// Pack weights f32 -> f16 into WMMA A-operand register order.
// W[dir][stage] is a 768x768 block matrix:
//   causal    : [ C | D ]   rows 0..511 -> y,   cols 0..255 -> state-in
//               [ A | B ]   rows 512..767 -> state-out, cols 256..767 -> u
//   anticausal: [ G | 0 ]
//               [ E | F ]
// Missing edge blocks are zero-filled.
// Tile (kt,mt): lane l owns 16 halfs: M = (l&15), K = {0..15}+16*(l>=16).
// ---------------------------------------------------------------------------
__global__ __launch_bounds__(256) void wprep_kernel(WPtrs p, _Float16* __restrict__ W) {
  const int r    = blockIdx.x * blockDim.x + threadIdx.x;
  const int lane = r & 31;
  const int tile = r >> 5;
  const int mt   = tile % MT_FULL;
  const int t2   = tile / MT_FULL;
  const int kt   = t2 % KT_FULL;
  const int t3   = t2 / KT_FULL;
  const int s    = t3 & 7;
  const int dir  = t3 >> 3;
  const int m    = mt * 16 + (lane & 15);
  const int khi  = (lane >> 4) << 4;

  v16h o;
  #pragma unroll
  for (int e = 0; e < 16; ++e) {
    const int K = e + khi;
    const int k = kt * 32 + K;
    float v = 0.0f;
    if (dir == 0) {                       // causal
      if (m < DOUT) {
        if (k < SDIM) { if (s > 0)            v = p.C[s][m * SDIM + k]; }
        else          {                       v = p.D[s][m * DI + (k - SDIM)]; }
      } else {
        const int ms = m - DOUT;
        if (k < SDIM) { if (s > 0 && s < 7)   v = p.A[s][ms * SDIM + k]; }
        else          { if (s < 7)            v = p.B[s][ms * DI + (k - SDIM)]; }
      }
    } else {                              // anticausal
      if (m < DOUT) {
        if (k < SDIM) { if (s < 7)            v = p.G[s][m * SDIM + k]; }
      } else {
        const int ms = m - DOUT;
        if (k < SDIM) { if (s > 0 && s < 7)   v = p.E[s][ms * SDIM + k]; }
        else          { if (s > 0)            v = p.F[s][ms * DI + (k - SDIM)]; }
      }
    }
    o[e] = (_Float16)v;
  }
  *(v16h*)(W + (size_t)tile * TILE_HALFS + lane * 16) = o;
}

// ---------------------------------------------------------------------------
// One GEMM segment: QN mtiles per wave (stride 8, base MT0), kt in [kt0,kt1),
// B operand from xsrc (B-packed LDS; xofs = 0 for state, 8 for u).
// No data-dependent branches inside.
// ---------------------------------------------------------------------------
template <int QN, int MT0>
__device__ __forceinline__ void mma_seg(
    v8f (*acc)[NT], const _Float16* __restrict__ Ws,
    const _Float16* __restrict__ xsrc, int kt0, int kt1, int xofs,
    int wave, int lane, int lhi, int llo) {
  #pragma unroll 2
  for (int kt = kt0; kt < kt1; ++kt) {
    v16h b[NT];
    #pragma unroll
    for (int nt = 0; nt < NT; ++nt)
      b[nt] = *(const v16h*)
          &xsrc[((2 * (kt - xofs) + lhi) * COLS + nt * 16 + llo) * 16];
    #pragma unroll
    for (int q = 0; q < QN; ++q) {
      const int mt = MT0 + wave + q * 8;
      const _Float16* ap =
          Ws + ((size_t)kt * MT_FULL + mt) * TILE_HALFS + lane * 16;
      if (kt + 1 < KT_FULL)                              // global_prefetch_b8
        __builtin_prefetch(ap + (size_t)MT_FULL * TILE_HALFS, 0, 3);
      const v16h a = *(const v16h*)ap;                   // 2x global_load_b128
      #pragma unroll
      for (int nt = 0; nt < NT; ++nt)
        acc[q][nt] = __builtin_amdgcn_wmma_f32_16x16x32_f16(
            false, a, false, b[nt], (short)0, acc[q][nt], false, false);
    }
  }
}

// y rows -> global (write + bias, or accumulate)
template <int QN, bool ADD>
__device__ __forceinline__ void epi_y(
    const v8f (*acc)[NT], float* __restrict__ out, const float* __restrict__ bias,
    int stg, int b0, int wave, int lhi, int llo) {
  #pragma unroll
  for (int q = 0; q < QN; ++q) {
    const int mt = wave + q * 8;
    const int f  = stg * DOUT + mt * 16 + 8 * lhi;       // 8 contiguous f / lane
    #pragma unroll
    for (int nt = 0; nt < NT; ++nt) {
      float* dst = out + (size_t)(b0 + nt * 16 + llo) * INSZ + f;
      if (ADD) {
        const v8f o = *(const v8f*)dst;                  // same wave wrote it
        *(v8f*)dst = o + acc[q][nt];
      } else {
        const v8f bb = *(const v8f*)(bias + f);
        *(v8f*)dst = acc[q][nt] + bb;
      }
    }
  }
}

// state rows -> LDS (B-operand pack), packed half2 stores
template <int QN>
__device__ __forceinline__ void epi_state(
    const v8f (*acc)[NT], _Float16* __restrict__ st_wr,
    int wave, int lhi, int llo) {
  #pragma unroll
  for (int q = 0; q < QN; ++q) {
    const int kc = wave + q * 8;                         // (mt-32): 16-K chunk
    #pragma unroll
    for (int nt = 0; nt < NT; ++nt) {
      const int n = nt * 16 + llo;
      #pragma unroll
      for (int rr = 0; rr < 8; rr += 2) {
        v2h h;
        h[0] = (_Float16)acc[q][nt][rr];
        h[1] = (_Float16)acc[q][nt][rr + 1];
        *(v2h*)&st_wr[(kc * COLS + n) * 16 + (rr + 8 * lhi)] = h;
      }
    }
  }
}

// ---------------------------------------------------------------------------
// Main scan kernel. One workgroup = 64 batch columns, 8 waves.
// pass 0: causal scan, writes out = y_c + bias.
// pass 1: anticausal scan, out += y_a.
// LDS: u tile (512x64 f16, B-packed, 64KB) + 2x state (256x64 f16, 32KB each).
// ---------------------------------------------------------------------------
__global__ __launch_bounds__(256) void sslayer_kernel(
    const float* __restrict__ U, const _Float16* __restrict__ W,
    const float* __restrict__ bias, float* __restrict__ out) {
  extern __shared__ _Float16 smem[];
  _Float16* u_lds = smem;                       // 32*64*16 halfs = 64 KB
  _Float16* st0   = smem + 32 * COLS * 16;      // 16*64*16 halfs = 32 KB
  _Float16* st1   = st0 + 16 * COLS * 16;       // 32 KB

  const int tid  = threadIdx.x;
  const int wave = tid >> 5;
  const int lane = tid & 31;
  const int lhi  = lane >> 4;
  const int llo  = lane & 15;
  const int b0   = blockIdx.x * COLS;

  for (int pass = 0; pass < 2; ++pass) {
    for (int it = 0; it < N_STAGES; ++it) {
      const int stg = (pass == 0) ? it : (N_STAGES - 1 - it);

      // ---- stage u tile: global f32 -> LDS f16 in B-operand pack ----
      if (pass == 0 || stg > 0) {
        const int n  = tid >> 2;                // 0..63 column
        const int kq = tid & 3;                 // 128-k quarter
        const float4* src =
            (const float4*)(U + (size_t)(b0 + n) * INSZ + stg * DI + kq * 128);
        #pragma unroll
        for (int s2 = 0; s2 < 32; ++s2) {
          const float4 v = src[s2];
          const int k = kq * 128 + s2 * 4;
          v4h h;
          h[0] = (_Float16)v.x; h[1] = (_Float16)v.y;
          h[2] = (_Float16)v.z; h[3] = (_Float16)v.w;
          *(v4h*)&u_lds[((k >> 4) * COLS + n) * 16 + (k & 15)] = h;
        }
      }
      __syncthreads();

      const int rb = it & 1;
      const _Float16* st_rd = rb ? st1 : st0;
      _Float16*       st_wr = rb ? st0 : st1;

      const v8f zero8 = {0.f, 0.f, 0.f, 0.f, 0.f, 0.f, 0.f, 0.f};
      v8f acc[6][NT];
      #pragma unroll
      for (int q = 0; q < 6; ++q)
        #pragma unroll
        for (int nt = 0; nt < NT; ++nt) acc[q][nt] = zero8;

      const _Float16* Ws =
          W + (size_t)((pass * 8 + stg) * KT_FULL * MT_FULL) * TILE_HALFS;

      if (pass == 0) {
        if (stg == 0) {                                   // y=D u, s'=B u
          mma_seg<6, 0>(acc, Ws, u_lds, 8, 24, 8, wave, lane, lhi, llo);
          epi_y<4, false>(acc, out, bias, stg, b0, wave, lhi, llo);
          epi_state<2>(acc + 4, st_wr, wave, lhi, llo);
        } else if (stg < 7) {                             // full [C|D;A|B]
          mma_seg<6, 0>(acc, Ws, st_rd, 0, 8, 0, wave, lane, lhi, llo);
          mma_seg<6, 0>(acc, Ws, u_lds, 8, 24, 8, wave, lane, lhi, llo);
          epi_y<4, false>(acc, out, bias, stg, b0, wave, lhi, llo);
          epi_state<2>(acc + 4, st_wr, wave, lhi, llo);
        } else {                                          // y only: [C|D]
          mma_seg<4, 0>(acc, Ws, st_rd, 0, 8, 0, wave, lane, lhi, llo);
          mma_seg<4, 0>(acc, Ws, u_lds, 8, 24, 8, wave, lane, lhi, llo);
          epi_y<4, false>(acc, out, bias, stg, b0, wave, lhi, llo);
        }
      } else {
        if (stg == 7) {                                   // s' = F u only
          mma_seg<2, 32>(acc + 4, Ws, u_lds, 8, 24, 8, wave, lane, lhi, llo);
          epi_state<2>(acc + 4, st_wr, wave, lhi, llo);
        } else if (stg > 0) {                             // [G;E] s  +  F u
          mma_seg<6, 0>(acc, Ws, st_rd, 0, 8, 0, wave, lane, lhi, llo);
          mma_seg<2, 32>(acc + 4, Ws, u_lds, 8, 24, 8, wave, lane, lhi, llo);
          epi_y<4, true>(acc, out, bias, stg, b0, wave, lhi, llo);
          epi_state<2>(acc + 4, st_wr, wave, lhi, llo);
        } else {                                          // y_a = G s only
          mma_seg<4, 0>(acc, Ws, st_rd, 0, 8, 0, wave, lane, lhi, llo);
          epi_y<4, true>(acc, out, bias, stg, b0, wave, lhi, llo);
        }
      }
      __syncthreads();
    }
  }
}

extern "C" void kernel_launch(void* const* d_in, const int* in_sizes, int n_in,
                              void* d_out, int out_size, void* d_ws, size_t ws_size,
                              hipStream_t stream) {
  (void)in_sizes; (void)n_in; (void)out_size; (void)ws_size;
  const float* U = (const float*)d_in[0];
  WPtrs p;
  for (int s = 0; s < 8; ++s) {
    p.A[s] = (const float*)d_in[1 + s];
    p.B[s] = (const float*)d_in[9 + s];
    p.C[s] = (const float*)d_in[17 + s];
    p.D[s] = (const float*)d_in[25 + s];
    p.E[s] = (const float*)d_in[33 + s];
    p.F[s] = (const float*)d_in[41 + s];
    p.G[s] = (const float*)d_in[49 + s];
  }
  const float* bias = (const float*)d_in[57];
  _Float16* W = (_Float16*)d_ws;  // 2*8*24*48 tiles * 1KB = 18.9 MB scratch

  const int wtiles = 2 * 8 * KT_FULL * MT_FULL;                 // 18432 tiles
  wprep_kernel<<<dim3(wtiles * 32 / 256), dim3(256), 0, stream>>>(p, W);
  sslayer_kernel<<<dim3(BATCH / COLS), dim3(256), 131072, stream>>>(
      U, W, bias, (float*)d_out);
}
